// GraphMoEDropoutExperts_10101763080592
// MI455X (gfx1250) — matrix operand order repaired and checked by reading, not compiled, not run on hardware
//
#include <hip/hip_runtime.h>
#include <cstdint>
#include <cstddef>

// ---------------------------------------------------------------------------
// Types for CDNA5 WMMA (wave32): 16x16x32 bf16 -> f32
// ---------------------------------------------------------------------------
typedef __attribute__((ext_vector_type(16))) __bf16 v16bf;
typedef __attribute__((ext_vector_type(8)))  float  v8f;

union Frag { uint4 q[2]; v16bf v; };

static __device__ __forceinline__ float bf2f(unsigned short b) {
  return __uint_as_float(((uint32_t)b) << 16);
}
static __device__ __forceinline__ unsigned short f2bf(float f) {
  uint32_t u = __float_as_uint(f);
  return (unsigned short)((u + 0x7FFFu + ((u >> 16) & 1u)) >> 16);  // RNE
}

static __device__ __forceinline__ v8f wmma_bf16(v16bf a, v16bf b, v8f c) {
  // (neg_a, A, neg_b, B, c_mod, C, reuse_a, reuse_b)
  return __builtin_amdgcn_wmma_f32_16x16x32_bf16(false, a, false, b, (short)0, c,
                                                 false, false);
}

// ---------------------------------------------------------------------------
// Utility kernels
// ---------------------------------------------------------------------------
__global__ void k_zero32(uint32_t* p, int n) {
  int i = blockIdx.x * blockDim.x + threadIdx.x;
  if (i < n) p[i] = 0u;
}
__global__ void k_copy32(const uint32_t* __restrict__ s, uint32_t* __restrict__ d, int n) {
  int i = blockIdx.x * blockDim.x + threadIdx.x;
  if (i < n) d[i] = s[i];
}
__global__ void k_f2bf(const float* __restrict__ s, unsigned short* __restrict__ d, int n) {
  int i = blockIdx.x * blockDim.x + threadIdx.x;
  if (i < n) d[i] = f2bf(s[i]);
}

// ---------------------------------------------------------------------------
// Encoder: h = relu(x @ enc_W + enc_b), [N,6]@[6,256] -> bf16 [N,256]
// ---------------------------------------------------------------------------
__global__ void k_encoder(const float* __restrict__ x, const float* __restrict__ W,
                          const float* __restrict__ b, unsigned short* __restrict__ h,
                          int N) {
  int i = blockIdx.x * blockDim.x + threadIdx.x;
  if (i >= N * 256) return;
  int n = i >> 8, c = i & 255;
  float s = b[c];
#pragma unroll
  for (int f = 0; f < 6; f++) s += x[n * 6 + f] * W[f * 256 + c];
  h[i] = f2bf(fmaxf(s, 0.f));
}

// ---------------------------------------------------------------------------
// Graph structure: degree / per-graph counts / CSR build
// ---------------------------------------------------------------------------
__global__ void k_count_edges(const int* __restrict__ ei, const int* __restrict__ batch,
                              int* deg, int* gedge, int E) {
  int i = blockIdx.x * blockDim.x + threadIdx.x;
  if (i >= E) return;
  atomicAdd(&deg[ei[E + i]], 1);          // dst in-degree
  atomicAdd(&gedge[batch[ei[i]]], 1);     // edges per graph (keyed by batch[src])
}
__global__ void k_count_nodes(const int* __restrict__ batch, int* gnode, int N) {
  int i = blockIdx.x * blockDim.x + threadIdx.x;
  if (i < N) atomicAdd(&gnode[batch[i]], 1);
}

__global__ void k_scan_block(const int* __restrict__ in, int* __restrict__ out,
                             int* __restrict__ bsum, int n) {
  __shared__ int s[256];
  int i = blockIdx.x * 256 + threadIdx.x;
  int x = (i < n) ? in[i] : 0;
  s[threadIdx.x] = x;
  __syncthreads();
  for (int off = 1; off < 256; off <<= 1) {
    int v = (threadIdx.x >= (unsigned)off) ? s[threadIdx.x - off] : 0;
    __syncthreads();
    s[threadIdx.x] += v;
    __syncthreads();
  }
  if (i < n) out[i] = s[threadIdx.x] - x;       // exclusive within block
  if (threadIdx.x == 255) bsum[blockIdx.x] = s[255];
}
__global__ void k_scan_top(int* b, int nb) {   // single block, nb <= 256
  __shared__ int s[256];
  int x = (threadIdx.x < (unsigned)nb) ? b[threadIdx.x] : 0;
  s[threadIdx.x] = x;
  __syncthreads();
  for (int off = 1; off < 256; off <<= 1) {
    int v = (threadIdx.x >= (unsigned)off) ? s[threadIdx.x - off] : 0;
    __syncthreads();
    s[threadIdx.x] += v;
    __syncthreads();
  }
  if (threadIdx.x < (unsigned)nb) b[threadIdx.x] = s[threadIdx.x] - x;
}
__global__ void k_scan_add(int* out, const int* __restrict__ bsum, int n) {
  int i = blockIdx.x * 256 + threadIdx.x;
  if (i < n) out[i] += bsum[i >> 8];
}
__global__ void k_settail(int* rowptr, int N, int E) { rowptr[N] = E; }

__global__ void k_fill(const int* __restrict__ ei, int* cursor, int* colidx, int E) {
  int i = blockIdx.x * blockDim.x + threadIdx.x;
  if (i >= E) return;
  int dst = ei[E + i];
  int slot = atomicAdd(&cursor[dst], 1);
  colidx[slot] = ei[i];
}

__global__ void k_sizefeat(const int* __restrict__ batch, const int* __restrict__ gnode,
                           const int* __restrict__ gedge, float* __restrict__ sf, int N) {
  int i = blockIdx.x * blockDim.x + threadIdx.x;
  if (i >= N) return;
  int g = batch[i];
  sf[2 * i + 0] = logf((float)gnode[g] + 1.f);
  sf[2 * i + 1] = logf((float)gedge[g] + 1.f);
}

// ---------------------------------------------------------------------------
// Aggregation: agg[n,:] = sum over in-edges of he[src,:]  (bf16 in/out, f32 acc)
// One block per node, 256 threads = 256 features (coalesced). L2-resident.
// ---------------------------------------------------------------------------
__global__ __launch_bounds__(256) void k_aggregate(const unsigned short* __restrict__ he,
                                                   const int* __restrict__ rowptr,
                                                   const int* __restrict__ colidx,
                                                   unsigned short* __restrict__ agg, int N) {
  int node = blockIdx.x;
  if (node >= N) return;
  int c = threadIdx.x;
  int beg = rowptr[node], end = rowptr[node + 1];
  float s = 0.f;
  for (int i = beg; i < end; i++) {
    s += bf2f(he[(size_t)colidx[i] * 256 + c]);
  }
  agg[(size_t)node * 256 + c] = f2bf(s);
}

// ---------------------------------------------------------------------------
// WMMA GEMM: D[nrows,256] = act(A@W1 [+ Agg@W2] + bias [+ sf@W1[256:258]])
// K = 256, Ncols = 256. Block = 8 waves, each wave = 32x32 tile, block = 128x64.
// Fragment layouts per CDNA5 ISA 7.12.2 (wave32).
// Out-of-range rows are CLAMPED (not predicated): the loads stay unconditional
// (no EXEC churn / no per-iteration zero-fills) and the store guard discards
// the clamped rows' results.
// ---------------------------------------------------------------------------
static __device__ __forceinline__ Frag load_a(const unsigned short* __restrict__ pa,
                                              int k0) {
  Frag f;
  const uint4* p = (const uint4*)(pa + k0);
  f.q[0] = p[0];   // K = kb .. kb+7
  f.q[1] = p[2];   // K = kb+16 .. kb+23
  return f;
}
static __device__ __forceinline__ Frag load_b(const unsigned short* __restrict__ pw,
                                              int k0) {
  Frag f;
  const unsigned short* p = pw + (size_t)k0 * 256;
  unsigned short t[16];
#pragma unroll
  for (int s = 0; s < 8; s++) {
    t[s] = p[(size_t)s * 256];
    t[8 + s] = p[(size_t)(16 + s) * 256];
  }
  __builtin_memcpy(&f, t, 32);
  return f;
}

static __device__ __forceinline__ void store_sub(const v8f& cc, int r0, int c0, int half,
                                                 int mr, const float* __restrict__ bias,
                                                 const float* __restrict__ sfeat,
                                                 const float* __restrict__ sfW,
                                                 unsigned short* __restrict__ out,
                                                 int nrows, bool relu, bool hassf) {
  int col = c0 + mr;
#pragma unroll
  for (int g = 0; g < 8; g++) {
    int r = r0 + g + 8 * half;
    if (r < nrows) {
      float v = cc[g] + bias[col];
      if (hassf) v += sfeat[2 * r] * sfW[col] + sfeat[2 * r + 1] * sfW[256 + col];
      if (relu) v = fmaxf(v, 0.f);
      out[(size_t)r * 256 + col] = f2bf(v);
    }
  }
}

template <bool HAS2, bool HASSF, bool RELU>
__global__ __launch_bounds__(256) void k_gemm256(const unsigned short* __restrict__ A,
                                                 const unsigned short* __restrict__ W1,
                                                 const unsigned short* __restrict__ Agg,
                                                 const unsigned short* __restrict__ W2,
                                                 const float* __restrict__ bias,
                                                 const float* __restrict__ sfeat,
                                                 const float* __restrict__ sfW,
                                                 unsigned short* __restrict__ out,
                                                 int nrows) {
  const int lane = threadIdx.x & 31;
  const int wave = threadIdx.x >> 5;
  const int wr = wave >> 1, wc = wave & 1;
  const int row0 = blockIdx.x * 128 + wr * 32;
  const int col0 = blockIdx.y * 64 + wc * 32;
  const int half = lane >> 4, mr = lane & 15;

  // clamped A row addresses (loop-invariant, unconditional loads)
  const int ar0 = min(row0 + mr, nrows - 1);
  const int ar1 = min(row0 + 16 + mr, nrows - 1);
  const int bk = 8 * half;    // K offset within tile for B-frags

  const unsigned short* pa0 = A + (size_t)ar0 * 256 + half * 8;
  const unsigned short* pa1 = A + (size_t)ar1 * 256 + half * 8;
  const unsigned short* pg0 = HAS2 ? Agg + (size_t)ar0 * 256 + half * 8 : nullptr;
  const unsigned short* pg1 = HAS2 ? Agg + (size_t)ar1 * 256 + half * 8 : nullptr;
  const unsigned short* pw1a = W1 + (size_t)bk * 256 + col0 + mr;
  const unsigned short* pw1b = W1 + (size_t)bk * 256 + col0 + 16 + mr;
  const unsigned short* pw2a = HAS2 ? W2 + (size_t)bk * 256 + col0 + mr : nullptr;
  const unsigned short* pw2b = HAS2 ? W2 + (size_t)bk * 256 + col0 + 16 + mr : nullptr;

  v8f c00 = {}, c01 = {}, c10 = {}, c11 = {};

#pragma unroll 2
  for (int k0 = 0; k0 < 256; k0 += 32) {
    Frag a0 = load_a(pa0, k0);
    Frag a1 = load_a(pa1, k0);
    Frag b0 = load_b(pw1a, k0);
    Frag b1 = load_b(pw1b, k0);
    c00 = wmma_bf16(a0.v, b0.v, c00);
    c01 = wmma_bf16(a0.v, b1.v, c01);
    c10 = wmma_bf16(a1.v, b0.v, c10);
    c11 = wmma_bf16(a1.v, b1.v, c11);
    if (HAS2) {
      Frag g0 = load_a(pg0, k0);
      Frag g1 = load_a(pg1, k0);
      Frag w0 = load_b(pw2a, k0);
      Frag w1 = load_b(pw2b, k0);
      c00 = wmma_bf16(g0.v, w0.v, c00);
      c01 = wmma_bf16(g0.v, w1.v, c01);
      c10 = wmma_bf16(g1.v, w0.v, c10);
      c11 = wmma_bf16(g1.v, w1.v, c11);
    }
  }

  store_sub(c00, row0,      col0,      half, mr, bias, sfeat, sfW, out, nrows, RELU, HASSF);
  store_sub(c01, row0,      col0 + 16, half, mr, bias, sfeat, sfW, out, nrows, RELU, HASSF);
  store_sub(c10, row0 + 16, col0,      half, mr, bias, sfeat, sfW, out, nrows, RELU, HASSF);
  store_sub(c11, row0 + 16, col0 + 16, half, mr, bias, sfeat, sfW, out, nrows, RELU, HASSF);
}

// ---------------------------------------------------------------------------
// Router tail: logits = r@W2 + b2; top-2 softmax -> dense sparse mask [N,8]
// ---------------------------------------------------------------------------
__global__ void k_route(const unsigned short* __restrict__ r, const float* __restrict__ W2,
                        const float* __restrict__ b2, float* __restrict__ sparse, int N) {
  int n = blockIdx.x * blockDim.x + threadIdx.x;
  if (n >= N) return;
  float lg[8];
#pragma unroll
  for (int e = 0; e < 8; e++) lg[e] = b2[e];
  for (int c = 0; c < 256; c++) {
    float rv = bf2f(r[(size_t)n * 256 + c]);
#pragma unroll
    for (int e = 0; e < 8; e++) lg[e] += rv * W2[c * 8 + e];
  }
  // top-2 (ties -> lowest index, matching jax.lax.top_k)
  float best = -3.4e38f; int i0 = 0;
#pragma unroll
  for (int e = 0; e < 8; e++) if (lg[e] > best) { best = lg[e]; i0 = e; }
  float best2 = -3.4e38f; int i1 = 0;
#pragma unroll
  for (int e = 0; e < 8; e++) if (e != i0 && lg[e] > best2) { best2 = lg[e]; i1 = e; }
  float e1 = expf(best2 - best);
  float w0 = 1.f / (1.f + e1);
  float w1 = e1 / (1.f + e1);
#pragma unroll
  for (int e = 0; e < 8; e++) {
    float w = (e == i0) ? w0 : ((e == i1) ? w1 : 0.f);
    sparse[(size_t)n * 8 + e] = w;
  }
}

// ---------------------------------------------------------------------------
// Output layer + masked combine: out[n,:] += sparse[n,e] * (he@Wr + agg@Wn + b)
// Skips nodes not routed to expert e (top-2 sparsity).
// ---------------------------------------------------------------------------
__global__ void k_combine(const unsigned short* __restrict__ he,
                          const unsigned short* __restrict__ agg,
                          const float* __restrict__ Wr, const float* __restrict__ Wn,
                          const float* __restrict__ b2, const float* __restrict__ sparse,
                          int eidx, float* __restrict__ out, int N) {
  int n = blockIdx.x * blockDim.x + threadIdx.x;
  if (n >= N) return;
  float s = sparse[(size_t)n * 8 + eidx];
  if (s == 0.f) return;
  float a0 = b2[0], a1 = b2[1];
#pragma unroll 4
  for (int c = 0; c < 256; c++) {
    float hv = bf2f(he[(size_t)n * 256 + c]);
    float av = bf2f(agg[(size_t)n * 256 + c]);
    a0 += hv * Wr[c * 2 + 0] + av * Wn[c * 2 + 0];
    a1 += hv * Wr[c * 2 + 1] + av * Wn[c * 2 + 1];
  }
  out[n * 2 + 0] += s * a0;
  out[n * 2 + 1] += s * a1;
}

// ---------------------------------------------------------------------------
// Host orchestration
// ---------------------------------------------------------------------------
extern "C" void kernel_launch(void* const* d_in, const int* in_sizes, int n_in,
                              void* d_out, int out_size, void* d_ws, size_t ws_size,
                              hipStream_t stream) {
  (void)n_in; (void)out_size; (void)ws_size;
  const float* x     = (const float*)d_in[0];
  const int*   ei    = (const int*)d_in[1];
  const int*   batch = (const int*)d_in[2];
  const float* encW  = (const float*)d_in[3];
  const float* encB  = (const float*)d_in[4];
  const float* rtrW1 = (const float*)d_in[5];
  const float* rtrB1 = (const float*)d_in[6];
  const float* rtrW2 = (const float*)d_in[7];
  const float* rtrB2 = (const float*)d_in[8];
  const float* hidWr = (const float*)d_in[9];
  const float* hidWn = (const float*)d_in[10];
  const float* hidB  = (const float*)d_in[11];
  const float* outWr = (const float*)d_in[12];
  const float* outWn = (const float*)d_in[13];
  const float* outB  = (const float*)d_in[14];
  float* out = (float*)d_out;

  const int N = in_sizes[0] / 6;
  const int E = in_sizes[1] / 2;
  const int GM = 65536;                 // generous graph-id range
  const int HW = 3 * 8 * 256 * 256;     // hidden weight element count

  // --- carve workspace (256B aligned slices) ---
  size_t off = 0;
  auto carve = [&](size_t bytes) -> void* {
    off = (off + 255) & ~(size_t)255;
    void* p = (char*)d_ws + off;
    off += bytes;
    return p;
  };
  unsigned short* h_b  = (unsigned short*)carve((size_t)N * 256 * 2);
  unsigned short* aggh = (unsigned short*)carve((size_t)N * 256 * 2);
  unsigned short* heA  = (unsigned short*)carve((size_t)N * 256 * 2);
  unsigned short* heB  = (unsigned short*)carve((size_t)N * 256 * 2);
  unsigned short* aggT = (unsigned short*)carve((size_t)N * 256 * 2);
  unsigned short* w1b  = (unsigned short*)carve((size_t)65536 * 2);
  unsigned short* wrb  = (unsigned short*)carve((size_t)HW * 2);
  unsigned short* wnb  = (unsigned short*)carve((size_t)HW * 2);
  float* sparse = (float*)carve((size_t)N * 8 * 4);
  float* sf     = (float*)carve((size_t)N * 2 * 4);
  int* deg    = (int*)carve((size_t)N * 4);
  int* rowptr = (int*)carve((size_t)(N + 1) * 4);
  int* cursor = (int*)carve((size_t)N * 4);
  int* colidx = (int*)carve((size_t)E * 4);
  int* bsum   = (int*)carve(1024 * 4);
  int* gnode  = (int*)carve((size_t)GM * 4);
  int* gedge  = (int*)carve((size_t)GM * 4);

  const int T = 256;
  auto nb = [&](int n) { return (n + T - 1) / T; };

  // zero counters + output
  k_zero32<<<nb(N), T, 0, stream>>>((uint32_t*)deg, N);
  k_zero32<<<nb(GM), T, 0, stream>>>((uint32_t*)gnode, GM);
  k_zero32<<<nb(GM), T, 0, stream>>>((uint32_t*)gedge, GM);
  k_zero32<<<nb(N * 2), T, 0, stream>>>((uint32_t*)out, N * 2);

  // weight conversion f32 -> bf16 (router W1 rows 0..255, all hidden weights)
  k_f2bf<<<nb(65536), T, 0, stream>>>(rtrW1, w1b, 65536);
  k_f2bf<<<nb(HW), T, 0, stream>>>(hidWr, wrb, HW);
  k_f2bf<<<nb(HW), T, 0, stream>>>(hidWn, wnb, HW);

  // encoder
  k_encoder<<<nb(N * 256), T, 0, stream>>>(x, encW, encB, h_b, N);

  // graph structure: counts -> exclusive scan -> CSR fill
  k_count_edges<<<nb(E), T, 0, stream>>>(ei, batch, deg, gedge, E);
  k_count_nodes<<<nb(N), T, 0, stream>>>(batch, gnode, N);
  const int NB = nb(N);
  k_scan_block<<<NB, 256, 0, stream>>>(deg, rowptr, bsum, N);
  k_scan_top<<<1, 256, 0, stream>>>(bsum, NB);
  k_scan_add<<<NB, 256, 0, stream>>>(rowptr, bsum, N);
  k_settail<<<1, 1, 0, stream>>>(rowptr, N, E);
  k_copy32<<<nb(N), T, 0, stream>>>((const uint32_t*)rowptr, (uint32_t*)cursor, N);
  k_fill<<<nb(E), T, 0, stream>>>(ei, cursor, colidx, E);

  // per-node size features
  k_sizefeat<<<nb(N), T, 0, stream>>>(batch, gnode, gedge, sf, N);

  // router: r = relu([h, sf] @ rtr_W1 + b1)  (sf folded into epilogue), r -> heA
  dim3 gg((N + 127) / 128, 4);
  k_gemm256<false, true, true><<<gg, 256, 0, stream>>>(
      h_b, w1b, nullptr, nullptr, rtrB1, sf, rtrW1 + 256 * 256, heA, N);
  k_route<<<nb(N), T, 0, stream>>>(heA, rtrW2, rtrB2, sparse, N);

  // shared layer-0 aggregate (he == h for every expert)
  k_aggregate<<<N, 256, 0, stream>>>(h_b, rowptr, colidx, aggh, N);

  for (int e = 0; e < 8; e++) {
    const unsigned short* Wr0 = wrb + (size_t)(0 * 8 + e) * 65536;
    const unsigned short* Wn0 = wnb + (size_t)(0 * 8 + e) * 65536;
    const unsigned short* Wr1 = wrb + (size_t)(1 * 8 + e) * 65536;
    const unsigned short* Wn1 = wnb + (size_t)(1 * 8 + e) * 65536;
    const unsigned short* Wr2 = wrb + (size_t)(2 * 8 + e) * 65536;
    const unsigned short* Wn2 = wnb + (size_t)(2 * 8 + e) * 65536;

    k_gemm256<true, false, true><<<gg, 256, 0, stream>>>(
        h_b, Wr0, aggh, Wn0, hidB + (size_t)(0 * 8 + e) * 256, nullptr, nullptr, heA, N);
    k_aggregate<<<N, 256, 0, stream>>>(heA, rowptr, colidx, aggT, N);

    k_gemm256<true, false, true><<<gg, 256, 0, stream>>>(
        heA, Wr1, aggT, Wn1, hidB + (size_t)(1 * 8 + e) * 256, nullptr, nullptr, heB, N);
    k_aggregate<<<N, 256, 0, stream>>>(heB, rowptr, colidx, aggT, N);

    k_gemm256<true, false, true><<<gg, 256, 0, stream>>>(
        heB, Wr2, aggT, Wn2, hidB + (size_t)(2 * 8 + e) * 256, nullptr, nullptr, heA, N);
    k_aggregate<<<N, 256, 0, stream>>>(heA, rowptr, colidx, aggT, N);

    k_combine<<<nb(N), T, 0, stream>>>(heA, aggT, outWr + (size_t)e * 512,
                                       outWn + (size_t)e * 512, outB + (size_t)e * 2,
                                       sparse, e, out, N);
  }
}
